// GCN_18253611008865
// MI455X (gfx1250) — matrix operand order repaired and checked by reading, not compiled, not run on hardware
//
#include <hip/hip_runtime.h>
#include <hip/hip_bf16.h>
#include <stddef.h>

// ---------------------------------------------------------------------------
// Fused GCN + KAN pipeline for gfx1250 (MI455X), wave32, WMMA f16 16x16x32.
//
// All grouped GEMMs are computed TRANSPOSED (D[n,sample]) so that:
//  - the per-group B matrix (p1^T) is built directly in registers with
//    packed-f16 VALU (v_pk_fma_f16 / v_pk_max_num_f16),
//  - per-group bias vectors ride in the WMMA C operand (row-indexed -> lane),
//  - per-group reductions against phi_w3 / g_w3 are per-lane register sums,
//  - phi feeds g_w1 linearly, so phi^T chunks are the B operand of the next
//    WMMA with zero layout conversion beyond a tiny per-wave LDS buffer.
// Static A operands (phi_w2^T, g_w1^T padded 56->64, g_w2^T) are pre-packed
// into an A-matrix lane layout with contiguous k (lane<16: row=lane, K 0..15;
// lane>=16: row=lane-16, K 16..31; halves ascending) in d_ws.
// ---------------------------------------------------------------------------

typedef __attribute__((ext_vector_type(16))) _Float16 v16h;
typedef __attribute__((ext_vector_type(8)))  _Float16 hv8;
typedef __attribute__((ext_vector_type(8)))  float    v8f;

#define NB 7
#define FIN 16
#define NFEAT 56
#define KD 32

// ws layout (bytes):
#define WS_AHAT  0                              // 49 f32
#define WS_APHI  512                            // 56 * 2 * 512 halves
#define WS_AG1   (512 + 114688)                 // 4 tiles * 512 halves
#define WS_AG2   (512 + 114688 + 4096)          // 2 tiles * 512 halves
#define WS_P1    (512 + 114688 + 4096 + 2048)   // 56 * 64 halves (w1|b1 f16)

__device__ __forceinline__ v8f wmma16(v16h a, v16h b, v8f c) {
  return __builtin_amdgcn_wmma_f32_16x16x32_f16(
      false, a, false, b, (short)0, c, false, false);
}

__device__ __forceinline__ v16h ldb16(const _Float16* p) {  // 32B, 16B-aligned
  v16h r;
  ((uint4*)&r)[0] = ((const uint4*)p)[0];
  ((uint4*)&r)[1] = ((const uint4*)p)[1];
  return r;
}

__device__ __forceinline__ hv8 ldh8(const _Float16* p) {    // 16B-aligned
  return *(const hv8*)p;
}

__device__ __forceinline__ v8f ldv8(const float* p) {       // 16B-aligned
  v8f r;
  ((float4*)&r)[0] = *(const float4*)p;
  ((float4*)&r)[1] = *(const float4*)(p + 4);
  return r;
}

__device__ __forceinline__ void lds_fence() {
  asm volatile("s_wait_dscnt 0" ::: "memory");
}

__device__ __forceinline__ int klocal(int hi, int j) {
  return hi * 16 + j;
}

// ---------------------------------------------------------------------------
// Prep: normalized adjacency + pack transposed weights into A-layout tiles,
// plus f16-packed (w1|b1) records for the KAN input layer.
// ---------------------------------------------------------------------------
__global__ __launch_bounds__(256) void gcnkan_prep(
    const int* __restrict__ edge, const float* __restrict__ phi_w2,
    const float* __restrict__ g_w1, const float* __restrict__ g_w2,
    const float* __restrict__ phi_w1, const float* __restrict__ phi_b1,
    float* __restrict__ wsA, _Float16* __restrict__ wsAphi,
    _Float16* __restrict__ wsAg1, _Float16* __restrict__ wsAg2,
    _Float16* __restrict__ wsP1) {
  const int tid = threadIdx.x;
  if (tid == 0) {
    float A[49];
    #pragma unroll
    for (int i = 0; i < 49; ++i) A[i] = 0.f;
    for (int e = 0; e < 14; ++e)
      A[edge[14 + e] * 7 + edge[e]] = 1.f;     // A[dst,src]=1
    #pragma unroll
    for (int n = 0; n < 7; ++n) A[n * 7 + n] = fmaxf(A[n * 7 + n], 1.f);
    float dinv[7];
    #pragma unroll
    for (int n = 0; n < 7; ++n) {
      float sm = 0.f;
      #pragma unroll
      for (int m = 0; m < 7; ++m) sm += A[n * 7 + m];
      dinv[n] = rsqrtf(sm);
    }
    #pragma unroll
    for (int n = 0; n < 7; ++n)
      #pragma unroll
      for (int m = 0; m < 7; ++m)
        wsA[n * 7 + m] = dinv[n] * A[n * 7 + m] * dinv[m];
  }
  // Aphi: phi_w2^T  tiles (kg, t): row n = t*16+(lane&15), K = klocal
  for (int idx = tid; idx < 56 * 1024; idx += 256) {
    int j = idx & 15, lane = (idx >> 4) & 31, t = (idx >> 9) & 1, kg = idx >> 10;
    int n = t * 16 + (lane & 15);
    int K = klocal(lane >> 4, j);
    wsAphi[idx] = (_Float16)phi_w2[kg * 1024 + K * 32 + n];
  }
  // Ag1: g_w1^T, K (=kg) padded 56->64, tile order (t, kc)
  for (int idx = tid; idx < 2048; idx += 256) {
    int j = idx & 15, lane = (idx >> 4) & 31, kc = (idx >> 9) & 1, t = idx >> 10;
    int n = t * 16 + (lane & 15);
    int Kg = kc * 32 + klocal(lane >> 4, j);
    wsAg1[idx] = (Kg < 56) ? (_Float16)g_w1[Kg * 32 + n] : (_Float16)0.f;
  }
  // Ag2: g_w2^T, tiles (t)
  for (int idx = tid; idx < 1024; idx += 256) {
    int j = idx & 15, lane = (idx >> 4) & 31, t = (idx >> 9) & 1;
    int n = t * 16 + (lane & 15);
    int K = klocal(lane >> 4, j);
    wsAg2[idx] = (_Float16)g_w2[K * 32 + n];
  }
  // P1: per kg, per half-wave: [w1(16 f16) | b1(16 f16)]
  for (int idx = tid; idx < 56 * 64; idx += 256) {
    int h = idx & 15, part = (idx >> 4) & 1, hi = (idx >> 5) & 1, kg = idx >> 6;
    const float* src = part ? phi_b1 : phi_w1;
    wsP1[idx] = (_Float16)src[kg * KD + hi * 16 + h];
  }
}

// ---------------------------------------------------------------------------
// Main fused kernel: 256 threads (8 waves) = 256 samples per block.
// Wave w owns samples 32w..32w+31 (two 16-sample tiles) -> wave-local LDS.
// ---------------------------------------------------------------------------
__global__ __launch_bounds__(256) void gcnkan_main(
    const float* __restrict__ x,
    const float* __restrict__ gcn1_w, const float* __restrict__ gcn1_b,
    const float* __restrict__ gcn2_w, const float* __restrict__ gcn2_b,
    const float* __restrict__ phi_b2, const float* __restrict__ phi_w3,
    const float* __restrict__ phi_b3,
    const float* __restrict__ g_b1, const float* __restrict__ g_b2,
    const float* __restrict__ g_w3, const float* __restrict__ g_b3,
    const float* __restrict__ line_w1, const float* __restrict__ line_b1,
    const float* __restrict__ line_w2, const float* __restrict__ line_b2,
    const float* __restrict__ wsA,
    const _Float16* __restrict__ wsAphi, const _Float16* __restrict__ wsAg1,
    const _Float16* __restrict__ wsAg2, const _Float16* __restrict__ wsP1,
    float* __restrict__ out) {
  __shared__ __align__(16) _Float16 s_flat[256 * 60];  // flat per sample
  __shared__ __align__(16) _Float16 s_ws[8 * 2048];    // 4KB scratch per wave
  __shared__ float sA[49];

  const int tid = threadIdx.x;
  const int lane = tid & 31;
  const int wave = tid >> 5;
  if (tid < 49) sA[tid] = wsA[tid];
  __syncthreads();

  // ---------------- Phase A: per-sample GCN + linear head (VALU) -----------
  const int s = blockIdx.x * 256 + tid;
  {
    const float* xs = x + (size_t)s * (NB * FIN);
    float t1[NB][8];
    #pragma unroll
    for (int m = 0; m < NB; ++m) {
      float xr[FIN];
      #pragma unroll
      for (int f = 0; f < FIN; ++f) xr[f] = xs[m * FIN + f];
      #pragma unroll
      for (int c = 0; c < 8; ++c) {
        float a = 0.f;
        #pragma unroll
        for (int f = 0; f < FIN; ++f) a += xr[f] * gcn1_w[f * 8 + c];
        t1[m][c] = a;
      }
    }
    float h1[NB][8];
    #pragma unroll
    for (int n = 0; n < NB; ++n)
      #pragma unroll
      for (int c = 0; c < 8; ++c) {
        float a = gcn1_b[c];
        #pragma unroll
        for (int m = 0; m < NB; ++m) a += sA[n * 7 + m] * t1[m][c];
        h1[n][c] = fmaxf(a, 0.f);
      }
    float t2[NB][8];
    #pragma unroll
    for (int m = 0; m < NB; ++m)
      #pragma unroll
      for (int c = 0; c < 8; ++c) {
        float a = 0.f;
        #pragma unroll
        for (int i = 0; i < 8; ++i) a += h1[m][i] * gcn2_w[i * 8 + c];
        t2[m][c] = a;
      }
    #pragma unroll
    for (int n = 0; n < NB; ++n)
      #pragma unroll
      for (int c = 0; c < 8; ++c) {
        float a = gcn2_b[c];
        #pragma unroll
        for (int m = 0; m < NB; ++m) a += sA[n * 7 + m] * t2[m][c];
        s_flat[tid * 60 + n * 8 + c] = (_Float16)fmaxf(a, 0.f);
      }
  }
  float lin;
  {
    float acc[24];
    #pragma unroll
    for (int j = 0; j < 24; ++j) acc[j] = line_b1[j];
    for (int k = 0; k < NFEAT; ++k) {
      float fv = (float)s_flat[tid * 60 + k];
      #pragma unroll
      for (int j = 0; j < 24; ++j) acc[j] += fv * line_w1[k * 24 + j];
    }
    lin = line_b2[0];
    #pragma unroll
    for (int j = 0; j < 24; ++j) lin += fmaxf(acc[j], 0.f) * line_w2[j];
  }
  lds_fence();

  // ---------------- Phase B: transposed WMMA pipeline ----------------------
  const int col = lane & 15;   // sample column within tile
  const int hi = lane >> 4;    // half-wave
  const v8f z = {};
  const hv8 hz = {};
  _Float16* wavebase = s_ws + wave * 2048;
  _Float16* ppart = wavebase + (hi * 16 + col) * 64;  // phi partials (f16)

  // hoisted per-lane g-MLP constants (n / o index = hi*8+j, +16 for 2nd tile)
  const v8f gb1A = ldv8(g_b1 + hi * 8), gb1B = ldv8(g_b1 + 16 + hi * 8);
  const v8f gb2A = ldv8(g_b2 + hi * 8), gb2B = ldv8(g_b2 + 16 + hi * 8);
  const v8f gw3A = ldv8(g_w3 + hi * 8), gw3B = ldv8(g_w3 + 16 + hi * 8);
  const float gb3v = g_b3[0];

  float kan0 = 0.f, kan1 = 0.f;

  for (int t = 0; t < 2; ++t) {
    const int ts = wave * 32 + t * 16;
    const _Float16* frow = s_flat + (ts + col) * 60;
    *(hv8*)(ppart + 56) = hz;   // pad kg 56..63

    // --- 56 grouped KAN GEMMs: p2^T[n,sample] = phi_w2^T @ p1^T + b2 -------
    #pragma unroll 1
    for (int kq = 0; kq < 14; ++kq) {
      uint2 fw = *(const uint2*)(frow + kq * 4);   // 4 flat values (f16)
      _Float16 f4v[4];
      __builtin_memcpy(f4v, &fw, 8);
      _Float16 tph[4];
      #pragma unroll
      for (int r = 0; r < 4; ++r) {
        const int kg = kq * 4 + r;
        const _Float16 fvh = f4v[r];
        // B = p1^T in registers (packed f16): lane = sample col, K = hi*16+j
        const _Float16* pp = wsP1 + (size_t)kg * 64 + hi * 32;
        hv8 w1a = ldh8(pp);       // w1[hi*16 + 0..7]
        hv8 b1a = ldh8(pp + 16);  // b1[hi*16 + 0..7]
        hv8 w1b = ldh8(pp + 8);   // w1[hi*16 + 8..15]
        hv8 b1b = ldh8(pp + 24);  // b1[hi*16 + 8..15]
        v16h bmat;
        ((hv8*)&bmat)[0] = __builtin_elementwise_max(w1a * fvh + b1a, hz);
        ((hv8*)&bmat)[1] = __builtin_elementwise_max(w1b * fvh + b1b, hz);
        v16h A0 = ldb16(wsAphi + (size_t)kg * 1024 + lane * 16);
        v16h A1 = ldb16(wsAphi + (size_t)kg * 1024 + 512 + lane * 16);
        // bias rides in the C operand (row n -> this lane's C-layout rows)
        v8f cb0 = ldv8(phi_b2 + kg * KD + hi * 8);        // n = hi*8+j
        v8f cb1 = ldv8(phi_b2 + kg * KD + 16 + hi * 8);   // n = 16+hi*8+j
        v8f d0 = wmma16(A0, bmat, cb0);
        v8f d1 = wmma16(A1, bmat, cb1);
        // phi partial (per-lane register reduction over this lane's 16 n's)
        v8f w3A = ldv8(phi_w3 + kg * KD + hi * 8);
        v8f w3B = ldv8(phi_w3 + kg * KD + 16 + hi * 8);
        float tacc = 0.5f * phi_b3[kg];   // each half adds b3/2 -> sum adds b3
        #pragma unroll
        for (int j = 0; j < 8; ++j) {
          tacc += fmaxf(d0[j], 0.f) * w3A[j];
          tacc += fmaxf(d1[j], 0.f) * w3B[j];
        }
        tph[r] = (_Float16)tacc;
      }
      __builtin_memcpy(ppart + kq * 4, tph, 8);   // one b64 store per 4 groups
    }
    lds_fence();

    // --- g1^T = g_w1^T @ phi^T + g_b1 (bias in C of first chunk) -----------
    v8f e0 = gb1A, e1 = gb1B;
    #pragma unroll
    for (int kc = 0; kc < 2; ++kc) {
      const int off = kc * 32 + hi * 16;
      const _Float16* q0 = wavebase + col * 64 + off;
      const _Float16* q1 = wavebase + (16 + col) * 64 + off;
      hv8 s0 = *(const hv8*)q0 + *(const hv8*)q1;        // v_pk_add_f16
      hv8 s1 = *(const hv8*)(q0 + 8) + *(const hv8*)(q1 + 8);
      v16h Bc;
      ((hv8*)&Bc)[0] = s0;
      ((hv8*)&Bc)[1] = s1;
      e0 = wmma16(ldb16(wsAg1 + (0 * 2 + kc) * 512 + lane * 16), Bc, e0);
      e1 = wmma16(ldb16(wsAg1 + (1 * 2 + kc) * 512 + lane * 16), Bc, e1);
    }
    // relu, pack g1^T to LDS (overlay scratch; DS in-order makes safe)
    hv8 P0, P1;
    #pragma unroll
    for (int j = 0; j < 8; ++j) {
      P0[j] = (_Float16)fmaxf(e0[j], 0.f);   // n = hi*8+j
      P1[j] = (_Float16)fmaxf(e1[j], 0.f);   // n = 16+hi*8+j
    }
    _Float16* gb = wavebase + col * 32;
    *(hv8*)(gb + hi * 8) = P0;
    *(hv8*)(gb + 16 + hi * 8) = P1;
    lds_fence();

    // --- g2^T = g_w2^T @ g1^T + g_b2 (bias in C) ---------------------------
    const _Float16* g2p = wavebase + col * 32 + hi * 16;
    v16h B2;
    ((hv8*)&B2)[0] = *(const hv8*)(g2p);
    ((hv8*)&B2)[1] = *(const hv8*)(g2p + 8);
    v8f f0 = wmma16(ldb16(wsAg2 + lane * 16), B2, gb2A);        // o = hi*8+j
    v8f f1 = wmma16(ldb16(wsAg2 + 512 + lane * 16), B2, gb2B);  // o = 16+hi*8+j
    // kan = relu(.) @ g_w3 + b3 : per-lane sum + one half-wave exchange
    float ks = 0.f;
    #pragma unroll
    for (int j = 0; j < 8; ++j) {
      ks += fmaxf(f0[j], 0.f) * gw3A[j];
      ks += fmaxf(f1[j], 0.f) * gw3B[j];
    }
    float tot = ks + __shfl_xor(ks, 16, 32) + gb3v;
    if (t == 0) kan0 = tot; else kan1 = tot;
  }

  // thread tid's sample lives in tile (hi): lane<16 -> tile0, lane>=16 -> tile1
  const float kanv = hi ? kan1 : kan0;
  out[s] = 0.5f * (lin + kanv);
}

// ---------------------------------------------------------------------------
extern "C" void kernel_launch(void* const* d_in, const int* in_sizes, int n_in,
                              void* d_out, int out_size, void* d_ws, size_t ws_size,
                              hipStream_t stream) {
  const float* x = (const float*)d_in[0];
  const int* edge = (const int*)d_in[1];
  const float* gcn1_w = (const float*)d_in[2];
  const float* gcn1_b = (const float*)d_in[3];
  const float* gcn2_w = (const float*)d_in[4];
  const float* gcn2_b = (const float*)d_in[5];
  const float* phi_w1 = (const float*)d_in[6];
  const float* phi_b1 = (const float*)d_in[7];
  const float* phi_w2 = (const float*)d_in[8];
  const float* phi_b2 = (const float*)d_in[9];
  const float* phi_w3 = (const float*)d_in[10];
  const float* phi_b3 = (const float*)d_in[11];
  const float* g_w1 = (const float*)d_in[12];
  const float* g_b1 = (const float*)d_in[13];
  const float* g_w2 = (const float*)d_in[14];
  const float* g_b2 = (const float*)d_in[15];
  const float* g_w3 = (const float*)d_in[16];
  const float* g_b3 = (const float*)d_in[17];
  const float* line_w1 = (const float*)d_in[18];
  const float* line_b1 = (const float*)d_in[19];
  const float* line_w2 = (const float*)d_in[20];
  const float* line_b2 = (const float*)d_in[21];

  char* ws = (char*)d_ws;   // needs ~129 KB
  float* wsA = (float*)(ws + WS_AHAT);
  _Float16* wsAphi = (_Float16*)(ws + WS_APHI);
  _Float16* wsAg1 = (_Float16*)(ws + WS_AG1);
  _Float16* wsAg2 = (_Float16*)(ws + WS_AG2);
  _Float16* wsP1 = (_Float16*)(ws + WS_P1);

  gcnkan_prep<<<1, 256, 0, stream>>>(edge, phi_w2, g_w1, g_w2, phi_w1, phi_b1,
                                     wsA, wsAphi, wsAg1, wsAg2, wsP1);

  const int nblocks = 32768 / 256;  // 128
  gcnkan_main<<<nblocks, 256, 0, stream>>>(
      x, gcn1_w, gcn1_b, gcn2_w, gcn2_b,
      phi_b2, phi_w3, phi_b3,
      g_b1, g_b2, g_w3, g_b3,
      line_w1, line_b1, line_w2, line_b2,
      wsA, wsAphi, wsAg1, wsAg2, wsP1,
      (float*)d_out);
}
